// SetTransformerAggregation_16063177687465
// MI455X (gfx1250) — compile-verified, matrix operand order checked by reading, not run on hardware
//
#include <hip/hip_runtime.h>
#include <cmath>

// ---------------- CDNA5 (gfx1250, wave32) Set Transformer ----------------
// Projections: v_wmma_f32_16x16x32_bf16 tiled GEMMs (99% of FLOPs).
// Attention:   small per-(batch,head) LDS softmax kernel (<1% of FLOPs).

typedef __bf16 bf16_t;
typedef __attribute__((ext_vector_type(16))) __bf16 bf16x16;
typedef __attribute__((ext_vector_type(8)))  __bf16 bf16x8;
typedef __attribute__((ext_vector_type(4)))  __bf16 bf16x4;
typedef __attribute__((ext_vector_type(8)))  float  f32x8;

#define DIMF     512
#define B_GRAPHS 1024
#define S_MAX    64
#define N_HEADS  8
#define E_HEAD   64
#define N_IND    32

// ---------------- utility kernels ----------------
__global__ void k_zero_f32(float* p, long long n) {
    long long i = (long long)blockIdx.x * blockDim.x + threadIdx.x;
    if (i < n) p[i] = 0.0f;
}
__global__ void k_zero_i32(int* p, int n) {
    int i = blockIdx.x * blockDim.x + threadIdx.x;
    if (i < n) p[i] = 0;
}
// Transpose + convert: s [512][512] fp32 row-major (k x n) -> d [n][k] bf16
__global__ __launch_bounds__(256) void k_cvtT_bf16(const float* __restrict__ s,
                                                   bf16_t* __restrict__ d) {
    __shared__ float tile[32][33];
    const int tx = threadIdx.x & 31, ty = threadIdx.x >> 5;   // 32 x 8
    const int kb = blockIdx.y * 32, nb = blockIdx.x * 32;
    #pragma unroll
    for (int r = 0; r < 32; r += 8)
        tile[ty + r][tx] = s[(long long)(kb + ty + r) * DIMF + nb + tx];
    __syncthreads();
    #pragma unroll
    for (int r = 0; r < 32; r += 8)
        d[(long long)(nb + ty + r) * DIMF + kb + tx] = (bf16_t)tile[tx][ty + r];
}
__global__ void k_starts_cnt(const int* __restrict__ batch, int* __restrict__ starts,
                             int* __restrict__ cnt, int nv) {
    int i = blockIdx.x * blockDim.x + threadIdx.x;
    if (i < nv) {
        int g = batch[i];
        atomicAdd(&cnt[g], 1);
        if (i == 0 || batch[i - 1] != g) starts[g] = i;
    }
}
__global__ void k_scatter(const float* __restrict__ H, const int* __restrict__ batch,
                          const int* __restrict__ starts, float* __restrict__ X, int nv) {
    long long gid = (long long)blockIdx.x * blockDim.x + threadIdx.x;
    long long total = (long long)nv * DIMF;
    if (gid >= total) return;
    int row = (int)(gid >> 9);
    int c = (int)(gid & (DIMF - 1));
    int g = batch[row];
    int p = row - starts[g];
    if (p >= 0 && p < S_MAX)
        X[((long long)g * S_MAX + p) * DIMF + c] = H[(long long)row * DIMF + c];
}
__global__ void k_add(const float* __restrict__ a, const float* __restrict__ b,
                      float* __restrict__ o, int n) {
    int i = blockIdx.x * blockDim.x + threadIdx.x;
    if (i < n) o[i] = a[i] + b[i];
}

// ---------------- WMMA GEMM: Y[M,512] = epi(A[M,512] @ W[512,512] + bias) ----------------
// EPI==0: Y = acc + bias ; EPI==1: Y = resid + relu(acc + bias)
// Wt is the bf16 weight PRE-TRANSPOSED to [n][k]. LDS tiles both end up [row][k]
// so each lane's fragment is two contiguous 16-byte runs (ds_load_b128 x2):
//   lane (half, lr): elems 0..7  = tile[row][half*8 + 0..7]
//                    elems 8..15 = tile[row][16 + half*8 + 0..7]
template <int EPI>
__global__ __launch_bounds__(256) void k_gemm512(
    const float* __restrict__ A, const bf16_t* __restrict__ Wt,
    const float* __restrict__ bias, const float* __restrict__ resid,
    float* __restrict__ Y, int M)
{
    constexpr int TM = 128, TN = 64, TK = 32;
    __shared__ bf16_t As[TM][TK + 8];   // row stride 80 B (16-B aligned)
    __shared__ bf16_t Bs[TN][TK + 8];   // [n][k]

    const int tid  = threadIdx.x;
    const int lane = tid & 31;          // wave32 on gfx1250
    const int wave = tid >> 5;          // 8 waves
    const int wm   = (wave & 3) * 32;   // 4 waves along M
    const int wn   = (wave >> 2) * 32;  // 2 waves along N
    const int half = lane >> 4;
    const int lr   = lane & 15;
    const int m0   = blockIdx.y * TM;
    const int n0   = blockIdx.x * TN;
    const bool fullM = (m0 + TM <= M);

    f32x8 acc[2][2] = {};
    float4 aReg[4];     // A: TM*TK/4 = 1024 float4-chunks / 256 thr = 4 per thread
    bf16x4 bReg[2];     // B: TN*TK/4 = 512 chunks / 256 thr = 2 per thread

    auto loadA = [&](int k0) {
        #pragma unroll
        for (int t = 0; t < 4; ++t) {
            int i = tid + t * 256;
            int r = i >> 3, cq = i & 7;          // 8 float4 per 32-wide row
            int gr = m0 + r;
            if (fullM || gr < M)
                aReg[t] = *(const float4*)(A + (long long)gr * DIMF + k0 + cq * 4);
            else
                aReg[t] = make_float4(0.f, 0.f, 0.f, 0.f);
        }
    };
    auto loadB = [&](int k0) {
        #pragma unroll
        for (int t = 0; t < 2; ++t) {
            int i = tid + t * 256;
            int n = i >> 3, kq = i & 7;
            bReg[t] = *(const bf16x4*)(Wt + (long long)(n0 + n) * DIMF + k0 + kq * 4);
        }
    };
    auto storeTiles = [&]() {
        #pragma unroll
        for (int t = 0; t < 4; ++t) {
            int i = tid + t * 256;
            int r = i >> 3, cq = i & 7;
            bf16x4 c;
            c[0] = (bf16_t)aReg[t].x; c[1] = (bf16_t)aReg[t].y;
            c[2] = (bf16_t)aReg[t].z; c[3] = (bf16_t)aReg[t].w;
            *(bf16x4*)&As[r][cq * 4] = c;
        }
        #pragma unroll
        for (int t = 0; t < 2; ++t) {
            int i = tid + t * 256;
            int n = i >> 3, kq = i & 7;
            *(bf16x4*)&Bs[n][kq * 4] = bReg[t];
        }
    };
    auto load_frag = [&](const bf16_t* rowbase) -> bf16x16 {
        const bf16x8* rp = (const bf16x8*)rowbase;
        bf16x8 lo = rp[half];
        bf16x8 hi = rp[2 + half];
        bf16x16 f;
        #pragma unroll
        for (int e = 0; e < 8; ++e) { f[e] = lo[e]; f[e + 8] = hi[e]; }
        return f;
    };

    loadA(0); loadB(0);
    for (int k0 = 0; k0 < DIMF; k0 += TK) {
        storeTiles();
        __syncthreads();
        if (k0 + TK < DIMF) { loadA(k0 + TK); loadB(k0 + TK); }   // prefetch next tile

        bf16x16 afrag[2], bfrag[2];
        #pragma unroll
        for (int i = 0; i < 2; ++i) afrag[i] = load_frag(&As[wm + i * 16 + lr][0]);
        #pragma unroll
        for (int j = 0; j < 2; ++j) bfrag[j] = load_frag(&Bs[wn + j * 16 + lr][0]);

        #pragma unroll
        for (int i = 0; i < 2; ++i)
            #pragma unroll
            for (int j = 0; j < 2; ++j)
                acc[i][j] = __builtin_amdgcn_wmma_f32_16x16x32_bf16(
                    false, afrag[i], false, bfrag[j], (short)0, acc[i][j], false, false);
        __syncthreads();
    }

    // Epilogue: C/D layout: lanes 0-15 rows 0..7 (VGPR e), lanes 16-31 rows 8..15; col = lr
    if (fullM) {
        #pragma unroll
        for (int i = 0; i < 2; ++i) {
            #pragma unroll
            for (int j = 0; j < 2; ++j) {
                int col = n0 + wn + j * 16 + lr;
                float bv = bias[col];
                long long rbase = (long long)(m0 + wm + i * 16 + half * 8) * DIMF + col;
                float* yp = Y + rbase;
                const float* rp = resid + rbase;   // unused when EPI==0
                #pragma unroll
                for (int e = 0; e < 8; ++e) {
                    float v = acc[i][j][e] + bv;
                    if (EPI == 1) v = rp[(long long)e * DIMF] + fmaxf(v, 0.0f);
                    yp[(long long)e * DIMF] = v;
                }
            }
        }
    } else {
        #pragma unroll
        for (int i = 0; i < 2; ++i) {
            #pragma unroll
            for (int j = 0; j < 2; ++j) {
                int col = n0 + wn + j * 16 + lr;
                float bv = bias[col];
                #pragma unroll
                for (int e = 0; e < 8; ++e) {
                    int row = m0 + wm + i * 16 + half * 8 + e;
                    if (row < M) {
                        float v = acc[i][j][e] + bv;
                        if (EPI == 1)
                            v = resid[(long long)row * DIMF + col] + fmaxf(v, 0.0f);
                        Y[(long long)row * DIMF + col] = v;
                    }
                }
            }
        }
    }
}

// ---------------- attention: O = Q + softmax(Q K^T * scale, mask) V  (per batch, head) ---------
// Q has qStride elements per batch (0 => broadcast across batch). Lq,Lk <= 64, head dim 64.
__global__ __launch_bounds__(128) void k_attn(
    const float* __restrict__ Q, const float* __restrict__ Kp, const float* __restrict__ Vp,
    float* __restrict__ O, const int* __restrict__ cnt,
    int Lq, int Lk, long long qStride, float scale)
{
    __shared__ float Ks[64][E_HEAD + 1];
    __shared__ float Vs[64][E_HEAD + 1];
    __shared__ float Sc[64][64];

    const int b = blockIdx.x, h = blockIdx.y, tid = threadIdx.x;
    const float* qbase = Q + (long long)b * qStride + h * E_HEAD;

    for (int i = tid; i < Lk * E_HEAD; i += 128) {
        int r = i >> 6, c = i & 63;
        long long row = ((long long)b * Lk + r) * DIMF + h * E_HEAD + c;
        Ks[r][c] = Kp[row];
        Vs[r][c] = Vp[row];
    }
    __syncthreads();

    int valid = cnt ? min(cnt[b], Lk) : Lk;

    for (int i = tid; i < Lq * Lk; i += 128) {
        int qi = i / Lk, ki = i - qi * Lk;
        float s;
        if (ki < valid) {
            s = 0.0f;
            const float* qr = qbase + (long long)qi * DIMF;
            #pragma unroll 8
            for (int e = 0; e < E_HEAD; ++e) s += qr[e] * Ks[ki][e];
            s *= scale;
        } else {
            s = -1e30f;
        }
        Sc[qi][ki] = s;
    }
    __syncthreads();

    if (tid < Lq) {
        float mx = -3.4e38f;
        for (int ki = 0; ki < Lk; ++ki) mx = fmaxf(mx, Sc[tid][ki]);
        float sum = 0.0f;
        for (int ki = 0; ki < Lk; ++ki) {
            float e = __expf(Sc[tid][ki] - mx);
            Sc[tid][ki] = e;
            sum += e;
        }
        float inv = 1.0f / sum;
        for (int ki = 0; ki < Lk; ++ki) Sc[tid][ki] *= inv;
    }
    __syncthreads();

    for (int i = tid; i < Lq * E_HEAD; i += 128) {
        int qi = i >> 6, c = i & 63;
        float a = qbase[(long long)qi * DIMF + c];   // residual q
        #pragma unroll 8
        for (int ki = 0; ki < Lk; ++ki) a += Sc[qi][ki] * Vs[ki][c];
        O[((long long)b * Lq + qi) * DIMF + h * E_HEAD + c] = a;
    }
}

// ---------------- host orchestration ----------------
struct MabDev {
    const bf16_t *wq, *wk, *wv, *wo;
    const float  *bq, *bk, *bv, *bo;
};

extern "C" void kernel_launch(void* const* d_in, const int* in_sizes, int n_in,
                              void* d_out, int out_size, void* d_ws, size_t ws_size,
                              hipStream_t stream) {
    const float* H     = (const float*)d_in[0];
    const int*   batch = (const int*)d_in[1];
    const int    nv    = in_sizes[1];   // total nodes (65536)

    // ---- workspace carve ----
    char* base = (char*)d_ws;
    size_t off = 0;
    auto take = [&](size_t bytes) -> void* {
        void* p = base + off;
        off += (bytes + 255) & ~(size_t)255;
        return p;
    };
    bf16_t* wbpool = (bf16_t*)take(29ull * DIMF * DIMF * sizeof(bf16_t));
    float* X  = (float*)take((size_t)B_GRAPHS * S_MAX * DIMF * 4);
    float* Qb = (float*)take((size_t)B_GRAPHS * S_MAX * DIMF * 4);
    float* Kb = (float*)take((size_t)B_GRAPHS * S_MAX * DIMF * 4);
    float* Vb = (float*)take((size_t)B_GRAPHS * S_MAX * DIMF * 4);
    float* Ob = (float*)take((size_t)B_GRAPHS * S_MAX * DIMF * 4);
    float* Hm = (float*)take((size_t)B_GRAPHS * N_IND * DIMF * 4);
    float* Pp = (float*)take((size_t)B_GRAPHS * DIMF * 4);
    float* Pq = (float*)take((size_t)B_GRAPHS * DIMF * 4);
    float* qI = (float*)take((size_t)N_IND * DIMF * 4);   // broadcast q (induced pts / seed)
    int* cnt    = (int*)take(B_GRAPHS * 4);
    int* starts = (int*)take(B_GRAPHS * 4);

    // ---- convert all 512x512 weights to bf16, TRANSPOSED to [n][k] ----
    int wslot = 0;
    auto conv = [&](int idx) -> const bf16_t* {
        bf16_t* dst = wbpool + (size_t)wslot * DIMF * DIMF;
        ++wslot;
        k_cvtT_bf16<<<dim3(DIMF / 32, DIMF / 32), 256, 0, stream>>>((const float*)d_in[idx], dst);
        return dst;
    };
    auto mkmab = [&](int b0) {
        MabDev m;
        m.wq = conv(b0 + 0); m.bq = (const float*)d_in[b0 + 1];
        m.wk = conv(b0 + 2); m.bk = (const float*)d_in[b0 + 3];
        m.wv = conv(b0 + 4); m.bv = (const float*)d_in[b0 + 5];
        m.wo = conv(b0 + 6); m.bo = (const float*)d_in[b0 + 7];
        return m;
    };
    // d_in layout (insertion order): 0 H, 1 batch, 2 isab1.I, 3-10 isab1.mab0,
    // 11-18 isab1.mab1, 19 isab2.I, 20-27 isab2.mab0, 28-35 isab2.mab1,
    // 36 pma.S, 37-44 pma.mab, 45-52 sab1, 53-60 sab2, 61 fc_w, 62 fc_b
    MabDev i1m0 = mkmab(3),  i1m1 = mkmab(11);
    MabDev i2m0 = mkmab(20), i2m1 = mkmab(28);
    MabDev pm   = mkmab(37);
    MabDev s1   = mkmab(45), s2 = mkmab(53);
    const bf16_t* fcW = conv(61);
    const float*  fcB = (const float*)d_in[62];

    const float scale = 1.0f / sqrtf((float)DIMF);

    auto gemm = [&](const float* A, const bf16_t* W, const float* bias,
                    const float* resid, float* Y, int M, int epi) {
        dim3 grid(DIMF / 64, (M + 127) / 128);
        if (epi == 0)
            k_gemm512<0><<<grid, 256, 0, stream>>>(A, W, bias, resid, Y, M);
        else
            k_gemm512<1><<<grid, 256, 0, stream>>>(A, W, bias, resid, Y, M);
    };
    auto attn = [&](const float* Q, long long qStride, const float* K, const float* V,
                    float* O, const int* msk, int Lq, int Lk) {
        k_attn<<<dim3(B_GRAPHS, N_HEADS), 128, 0, stream>>>(Q, K, V, O, msk, Lq, Lk, qStride, scale);
    };

    // ---- ragged -> padded set ----
    long long nX = (long long)B_GRAPHS * S_MAX * DIMF;
    k_zero_f32<<<(unsigned)((nX + 255) / 256), 256, 0, stream>>>(X, nX);
    k_zero_i32<<<(B_GRAPHS + 255) / 256, 256, 0, stream>>>(cnt, B_GRAPHS);
    k_starts_cnt<<<(nv + 255) / 256, 256, 0, stream>>>(batch, starts, cnt, nv);
    long long nS = (long long)nv * DIMF;
    k_scatter<<<(unsigned)((nS + 255) / 256), 256, 0, stream>>>(H, batch, starts, X, nv);

    // ---- ISAB (x2): Hm = MAB(I, X, mask); X = MAB(X, Hm) ----
    auto run_isab = [&](const float* Ivec, const MabDev& m0, const MabDev& m1) {
        // mab0: Q = I broadcast (q projection is batch-invariant -> compute once)
        gemm(Ivec, m0.wq, m0.bq, nullptr, qI, N_IND, 0);
        gemm(X, m0.wk, m0.bk, nullptr, Kb, B_GRAPHS * S_MAX, 0);
        gemm(X, m0.wv, m0.bv, nullptr, Vb, B_GRAPHS * S_MAX, 0);
        attn(qI, 0, Kb, Vb, Ob, cnt, N_IND, S_MAX);
        gemm(Ob, m0.wo, m0.bo, Ob, Hm, B_GRAPHS * N_IND, 1);
        // mab1: Q = X, K = Hm, no mask
        gemm(X, m1.wq, m1.bq, nullptr, Qb, B_GRAPHS * S_MAX, 0);
        gemm(Hm, m1.wk, m1.bk, nullptr, Kb, B_GRAPHS * N_IND, 0);
        gemm(Hm, m1.wv, m1.bv, nullptr, Vb, B_GRAPHS * N_IND, 0);
        attn(Qb, (long long)S_MAX * DIMF, Kb, Vb, Ob, nullptr, S_MAX, N_IND);
        gemm(Ob, m1.wo, m1.bo, Ob, X, B_GRAPHS * S_MAX, 1);
    };
    run_isab((const float*)d_in[2],  i1m0, i1m1);
    run_isab((const float*)d_in[19], i2m0, i2m1);

    // ---- PMA: P = MAB(Seed, X, mask), 1 seed per graph ----
    gemm((const float*)d_in[36], pm.wq, pm.bq, nullptr, qI, 1, 0);
    gemm(X, pm.wk, pm.bk, nullptr, Kb, B_GRAPHS * S_MAX, 0);
    gemm(X, pm.wv, pm.bv, nullptr, Vb, B_GRAPHS * S_MAX, 0);
    attn(qI, 0, Kb, Vb, Ob, cnt, 1, S_MAX);
    gemm(Ob, pm.wo, pm.bo, Ob, Pp, B_GRAPHS, 1);

    // ---- SAB1/SAB2 on [B,1,512]: softmax over 1 key == 1 -> o = q + v ----
    gemm(Pp, s1.wq, s1.bq, nullptr, Qb, B_GRAPHS, 0);
    gemm(Pp, s1.wv, s1.bv, nullptr, Vb, B_GRAPHS, 0);
    k_add<<<(B_GRAPHS * DIMF + 255) / 256, 256, 0, stream>>>(Qb, Vb, Ob, B_GRAPHS * DIMF);
    gemm(Ob, s1.wo, s1.bo, Ob, Pq, B_GRAPHS, 1);

    gemm(Pq, s2.wq, s2.bq, nullptr, Qb, B_GRAPHS, 0);
    gemm(Pq, s2.wv, s2.bv, nullptr, Vb, B_GRAPHS, 0);
    k_add<<<(B_GRAPHS * DIMF + 255) / 256, 256, 0, stream>>>(Qb, Vb, Ob, B_GRAPHS * DIMF);
    gemm(Ob, s2.wo, s2.bo, Ob, Pp, B_GRAPHS, 1);

    // ---- final FC -> d_out [B, 512] ----
    gemm(Pp, fcW, fcB, nullptr, (float*)d_out, B_GRAPHS, 0);

    (void)n_in; (void)out_size; (void)ws_size; (void)in_sizes;
}
